// SparseSelfAttention_85220741087704
// MI455X (gfx1250) — compile-verified, matrix operand order checked by reading, not run on hardware
//
#include <hip/hip_runtime.h>
#include <hip/hip_bf16.h>

typedef __attribute__((ext_vector_type(16))) _Float16 v16h;
typedef __attribute__((ext_vector_type(8)))  _Float16 v8h;
typedef __attribute__((ext_vector_type(8)))  float    v8f;
typedef __attribute__((ext_vector_type(4)))  float    v4f;

#define N_TOK 2304
#define CH    256
#define NELEM (N_TOK * CH)

// CDNA5 16-bit A/B fragment (16x32): lane's 16 elements are two contiguous
// 8-half runs: k = 8g..8g+7 (e=0..7) and k = 16+8g..23+8g (e=8..15), g=lane/16.
// -> load each fragment as two 16-byte vector loads.
__device__ __forceinline__ v16h loadfrag(const _Float16* __restrict__ p, int g8) {
    v8h lo = *(const v8h*)(p + g8);
    v8h hi = *(const v8h*)(p + 16 + g8);
    v16h r;
#pragma unroll
    for (int i = 0; i < 8; ++i) { r[i] = lo[i]; r[8 + i] = hi[i]; }
    return r;
}

// Same fragment geometry but from an f32 source (vector load + packed convert).
__device__ __forceinline__ v16h loadfrag_f32(const float* __restrict__ p, int g8) {
    v4f a = *(const v4f*)(p + g8);
    v4f b = *(const v4f*)(p + g8 + 4);
    v4f c = *(const v4f*)(p + 16 + g8);
    v4f d = *(const v4f*)(p + 16 + g8 + 4);
    v16h r;
#pragma unroll
    for (int i = 0; i < 4; ++i) {
        r[i]      = (_Float16)a[i];
        r[4 + i]  = (_Float16)b[i];
        r[8 + i]  = (_Float16)c[i];
        r[12 + i] = (_Float16)d[i];
    }
    return r;
}

// ---------------------------------------------------------------------------
// Kernel 1: QKV projection. Block = 256 threads = 8 waves; each wave one 16x16
// output tile. grid = (N/16, 6): blockIdx.y -> {matrix 0..2} x {column half}.
// Q is pre-scaled by 1/SCALE = 1/16 (folds score scaling).
// ---------------------------------------------------------------------------
__global__ __launch_bounds__(256) void qkv_proj_kernel(
    const float* __restrict__ x,
    const float* __restrict__ Wq, const float* __restrict__ bq,
    const float* __restrict__ Wk, const float* __restrict__ bk,
    const float* __restrict__ Wv, const float* __restrict__ bv,
    _Float16* __restrict__ Qh, _Float16* __restrict__ Kh, float* __restrict__ Vf)
{
    const int lane = threadIdx.x & 31;
    const int wave = threadIdx.x >> 5;
    const int tile = blockIdx.x;                    // 16-row tile of tokens
    const int mat  = blockIdx.y >> 1;               // 0=Q 1=K 2=V
    const int c0   = (blockIdx.y & 1) * 128 + wave * 16;

    const float* W    = (mat == 0) ? Wq : (mat == 1) ? Wk : Wv;
    const float* bias = (mat == 0) ? bq : (mat == 1) ? bk : bv;

    const int l15 = lane & 15;
    const int g8  = (lane >> 4) << 3;
    const float* arow = x + (tile * 16 + l15) * CH; // token row (A fragment row)
    const float* brow = W + (c0 + l15) * CH;        // B[k][n] = W[n][k]

    v8f acc = {};
#pragma unroll
    for (int kb = 0; kb < CH; kb += 32) {
        v16h a = loadfrag_f32(arow + kb, g8);
        v16h b = loadfrag_f32(brow + kb, g8);
        acc = __builtin_amdgcn_wmma_f32_16x16x32_f16(false, a, false, b,
                                                     (short)0, acc, false, false);
    }
    const float bcol = bias[c0 + l15];
#pragma unroll
    for (int r = 0; r < 8; ++r) {
        int row = tile * 16 + r + g8;               // D: vgpr r -> row r+8g, col lane%16
        int col = c0 + l15;
        float v = acc[r] + bcol;
        if (mat == 0)       Qh[row * CH + col] = (_Float16)(v * 0.0625f); // /16
        else if (mat == 1)  Kh[row * CH + col] = (_Float16)v;
        else                Vf[row * CH + col] = v;
    }
}

// ---------------------------------------------------------------------------
// Kernel 2: VsT[c][n] = sum_{j in {-4,-2,0,2,4}} V[(n+j) mod N][c], stored
// TRANSPOSED (token axis contiguous) so the P@Vs B-fragments are b128 loads.
// ---------------------------------------------------------------------------
__global__ void vsum_kernel(const float* __restrict__ Vf, _Float16* __restrict__ VsT)
{
    int idx = blockIdx.x * blockDim.x + threadIdx.x; // flat over [c][n], n fastest
    if (idx >= NELEM) return;
    int c = idx / N_TOK, n = idx - c * N_TOK;
    float s = 0.f;
#pragma unroll
    for (int o = -4; o <= 4; o += 2) {
        int nn = n + o;
        if (nn < 0) nn += N_TOK;
        if (nn >= N_TOK) nn -= N_TOK;
        s += Vf[nn * CH + c];
    }
    VsT[idx] = (_Float16)s;
}

// ---------------------------------------------------------------------------
// Kernel 3: attention. grid = (N/128, 4 heads); block = 256 = 8 waves; each
// wave owns 16 query rows of head hq, loops over the 4 kh partners, runs a
// flash-attention pass of 2304 keys in 32-key steps, and accumulates
// weight * softmax(Q Khᵀ) @ Vs[vh] into registers. Deterministic plain stores.
// ---------------------------------------------------------------------------
__global__ __launch_bounds__(256) void attn_kernel(
    const _Float16* __restrict__ Qh, const _Float16* __restrict__ Kh,
    const _Float16* __restrict__ VsT, float* __restrict__ out)
{
    __shared__ __align__(16) _Float16 lds[8][16 * 32]; // per-wave P (16 rows x 32 keys)

    const int lane = threadIdx.x & 31;
    const int wave = threadIdx.x >> 5;
    const int hq   = blockIdx.y;
    const int qbase = blockIdx.x * 128 + wave * 16;
    const int l15 = lane & 15;
    const int g8  = (lane >> 4) << 3;

    // Q fragments (d-chunks 0..31 and 32..63), shared across all kh partners
    const _Float16* qrow = Qh + (qbase + l15) * CH + hq * 64;
    const v16h qa0 = loadfrag(qrow, g8);
    const v16h qa1 = loadfrag(qrow + 32, g8);

    v8f f0 = {}, f1 = {}, f2 = {}, f3 = {};         // final accumulators (16 rows x 64 cols)

    for (int kh = 0; kh < 4; ++kh) {
        const int delta = (hq - kh) & 3;            // h mod 4
        const int vh    = (hq + delta) & 3;
        const float weight = (delta == 2) ? 2.0f : 1.0f;  // h=-2 and h=+2 coincide

        float mi[8], li[8];
#pragma unroll
        for (int r = 0; r < 8; ++r) { mi[r] = -1e30f; li[r] = 0.f; }
        v8f o0 = {}, o1 = {}, o2 = {}, o3 = {};

#pragma unroll 2
        for (int key0 = 0; key0 < N_TOK; key0 += 32) {
            // ---- S = Q Khᵀ (two 16x16 tiles, K=64 as 2x32) ----
            const _Float16* k0 = Kh + (key0 + l15) * CH + kh * 64;
            const _Float16* k1 = k0 + 16 * CH;
            const _Float16* vbase = VsT + (vh * 64 + l15) * N_TOK + key0;

            // prefetch two key-blocks ahead (L2-resident; hides L0 latency)
            if (key0 + 64 < N_TOK) {
                __builtin_prefetch(k0 + 64 * CH, 0, 3);
                __builtin_prefetch(k1 + 64 * CH, 0, 3);
                __builtin_prefetch(vbase + 64, 0, 3);
            }

            v16h b00 = loadfrag(k0, g8),      b01 = loadfrag(k0 + 32, g8);
            v16h b10 = loadfrag(k1, g8),      b11 = loadfrag(k1 + 32, g8);
            v8f s0 = {}, s1 = {};
            // interleave the two independent accumulation chains
            s0 = __builtin_amdgcn_wmma_f32_16x16x32_f16(false, qa0, false, b00, (short)0, s0, false, false);
            s1 = __builtin_amdgcn_wmma_f32_16x16x32_f16(false, qa0, false, b10, (short)0, s1, false, false);
            s0 = __builtin_amdgcn_wmma_f32_16x16x32_f16(false, qa1, false, b01, (short)0, s0, false, false);
            s1 = __builtin_amdgcn_wmma_f32_16x16x32_f16(false, qa1, false, b11, (short)0, s1, false, false);

            // ---- online softmax over this 32-key block (row-wise) ----
#pragma unroll
            for (int r = 0; r < 8; ++r) {
                float v = fmaxf(s0[r], s1[r]);
                v = fmaxf(v, __shfl_xor(v, 1));
                v = fmaxf(v, __shfl_xor(v, 2));
                v = fmaxf(v, __shfl_xor(v, 4));
                v = fmaxf(v, __shfl_xor(v, 8));     // masks <16: stays in lane half
                float mnew = fmaxf(mi[r], v);
                float corr = __expf(mi[r] - mnew);
                float p0 = __expf(s0[r] - mnew);
                float p1 = __expf(s1[r] - mnew);
                float rs = p0 + p1;
                rs += __shfl_xor(rs, 1);
                rs += __shfl_xor(rs, 2);
                rs += __shfl_xor(rs, 4);
                rs += __shfl_xor(rs, 8);
                li[r] = li[r] * corr + rs;
                mi[r] = mnew;
                o0[r] *= corr; o1[r] *= corr; o2[r] *= corr; o3[r] *= corr;
                lds[wave][(r + g8) * 32 + l15]      = (_Float16)p0;   // P[row][key]
                lds[wave][(r + g8) * 32 + 16 + l15] = (_Float16)p1;
            }
            asm volatile("s_wait_dscnt 0" ::: "memory");  // wave-private LDS RAW fence

            // ---- O += P @ Vs[vh]  (A = P from LDS, B = VsT rows, 4 col chunks) ----
            v16h pa = loadfrag(&lds[wave][l15 * 32], g8);  // two ds_load_b128
            v16h bv0 = loadfrag(vbase, g8);
            v16h bv1 = loadfrag(vbase + 16 * N_TOK, g8);
            v16h bv2 = loadfrag(vbase + 32 * N_TOK, g8);
            v16h bv3 = loadfrag(vbase + 48 * N_TOK, g8);
            o0 = __builtin_amdgcn_wmma_f32_16x16x32_f16(false, pa, false, bv0, (short)0, o0, false, false);
            o1 = __builtin_amdgcn_wmma_f32_16x16x32_f16(false, pa, false, bv1, (short)0, o1, false, false);
            o2 = __builtin_amdgcn_wmma_f32_16x16x32_f16(false, pa, false, bv2, (short)0, o2, false, false);
            o3 = __builtin_amdgcn_wmma_f32_16x16x32_f16(false, pa, false, bv3, (short)0, o3, false, false);
        }
        // fold this kh partner into the final accumulator
#pragma unroll
        for (int r = 0; r < 8; ++r) {
            float inv = weight / li[r];
            f0[r] += o0[r] * inv;
            f1[r] += o1[r] * inv;
            f2[r] += o2[r] * inv;
            f3[r] += o3[r] * inv;
        }
    }
    // out[n][hq*64 + c] ; exactly one writer per element -> plain stores
#pragma unroll
    for (int r = 0; r < 8; ++r) {
        int row = qbase + r + g8;
        float* orow = out + row * CH + hq * 64 + l15;
        orow[0]  = f0[r];
        orow[16] = f1[r];
        orow[32] = f2[r];
        orow[48] = f3[r];
    }
}

extern "C" void kernel_launch(void* const* d_in, const int* in_sizes, int n_in,
                              void* d_out, int out_size, void* d_ws, size_t ws_size,
                              hipStream_t stream)
{
    const float* x  = (const float*)d_in[0];
    const float* Wq = (const float*)d_in[1];
    const float* bq = (const float*)d_in[2];
    const float* Wk = (const float*)d_in[3];
    const float* bk = (const float*)d_in[4];
    const float* Wv = (const float*)d_in[5];
    const float* bv = (const float*)d_in[6];
    float* out = (float*)d_out;

    char* ws = (char*)d_ws;
    _Float16* Qh  = (_Float16*)(ws);                 // N*CH halves  (1.13 MB)
    _Float16* Kh  = (_Float16*)(ws + 1179648);       // N*CH halves
    float*    Vf  = (float*)   (ws + 2359296);       // N*CH floats  (2.25 MB)
    _Float16* VsT = (_Float16*)(ws + 4718592);       // N*CH halves, transposed [c][n]

    qkv_proj_kernel<<<dim3(N_TOK / 16, 6), 256, 0, stream>>>(
        x, Wq, bq, Wk, bk, Wv, bv, Qh, Kh, Vf);
    vsum_kernel<<<(NELEM + 255) / 256, 256, 0, stream>>>(Vf, VsT);
    attn_kernel<<<dim3(N_TOK / 128, 4), 256, 0, stream>>>(Qh, Kh, VsT, out);
}
